// ParallelRetention_84482006712633
// MI455X (gfx1250) — compile-verified, matrix operand order checked by reading, not run on hardware
//
#include <hip/hip_runtime.h>
#include <math.h>

#define NN 8192
#define DIM 1024
#define GN_EPS 1e-5f

typedef __attribute__((ext_vector_type(16))) __bf16 v16bf;
typedef __attribute__((ext_vector_type(8)))  float  v8f;

union FragU { uint4 u[2]; v16bf v; };

__device__ __forceinline__ v8f wmma_bf16(v16bf a, v16bf b, v8f c) {
  return __builtin_amdgcn_wmma_f32_16x16x32_bf16(false, a, false, b, (short)0, c, false, false);
}

// A/B fragment loader for v_wmma_f32_16x16x32_bf16.
// Lane l (l&15 = row/col index, l>>4 = half) holds K = {k0+h*8 .. +7} and {k0+16+h*8 .. +7}.
__device__ __forceinline__ v16bf load_frag_g(const __bf16* base, int row, int ld, int k0, int half) {
  const char* p = (const char*)base + (((size_t)row * (size_t)ld + (size_t)(k0 + half * 8)) * 2u);
  FragU f;
  f.u[0] = *(const uint4*)p;
  f.u[1] = *(const uint4*)(p + 32);
  return f.v;
}

// ---------------------------------------------------------------- converts
__global__ void f32_to_bf16_kernel(const float* __restrict__ in, __bf16* __restrict__ out, int n) {
  for (int i = blockIdx.x * blockDim.x + threadIdx.x; i < n; i += gridDim.x * blockDim.x)
    out[i] = (__bf16)in[i];
}

// ---------------------------------------------------------------- generic WMMA GEMM
// C[M,Nout] (+)= A[M,K](bf16) * W[Nout,ldw](bf16)^T  + bias, epilogue act.
// act: 0 none, 1 exact GELU, 2 PReLU(slope per channel)
// bfout: optional bf16 mirror; bf_transposed: write bfout as [Nout, M]
__global__ __launch_bounds__(256) void gemm_wmma(
    const __bf16* __restrict__ A, int M, int K,
    const __bf16* __restrict__ W, int ldw, int Nout,
    const float* __restrict__ bias, const float* __restrict__ slope,
    float* __restrict__ Cout, int accumulate, int act,
    __bf16* __restrict__ bfout, int bf_transposed)
{
  const int tid  = threadIdx.x;
  const int lane = tid & 31;
  const int wid  = tid >> 5;
  const int ln   = lane & 15;
  const int half = lane >> 4;
  const int wm   = wid >> 2;     // 0..1
  const int wn   = wid & 3;      // 0..3
  const int rb   = blockIdx.x * 128 + wm * 64;
  const int cb   = blockIdx.y * 128 + wn * 32;

  v8f c[4][2] = {};

  for (int k0 = 0; k0 < K; k0 += 32) {
    v16bf a[4], b[2];
#pragma unroll
    for (int i = 0; i < 4; i++) a[i] = load_frag_g(A, rb + i * 16 + ln, K, k0, half);
#pragma unroll
    for (int j = 0; j < 2; j++) b[j] = load_frag_g(W, cb + j * 16 + ln, ldw, k0, half);
#pragma unroll
    for (int i = 0; i < 4; i++)
#pragma unroll
      for (int j = 0; j < 2; j++)
        c[i][j] = wmma_bf16(a[i], b[j], c[i][j]);
  }

#pragma unroll
  for (int j = 0; j < 2; j++) {
    const int col = cb + j * 16 + ln;
    const float bj = bias ? bias[col] : 0.0f;
    const float sj = (act == 2 && slope) ? slope[col] : 0.0f;
#pragma unroll
    for (int i = 0; i < 4; i++) {
      alignas(16) __bf16 tmp[8];
      float vals[8];
#pragma unroll
      for (int r = 0; r < 8; r++) {
        const int row = rb + i * 16 + 8 * half + r;
        float v = c[i][j][r];
        if (accumulate) v += Cout[(size_t)row * Nout + col];
        v += bj;
        if (act == 1)      v = 0.5f * v * (1.0f + erff(v * 0.70710678118654752f));
        else if (act == 2) v = (v >= 0.0f) ? v : sj * v;
        vals[r] = v;
        if (Cout) Cout[(size_t)row * Nout + col] = v;
      }
      if (bfout) {
        if (bf_transposed) {
#pragma unroll
          for (int r = 0; r < 8; r++) tmp[r] = (__bf16)vals[r];
          *(uint4*)(bfout + (size_t)col * M + rb + i * 16 + 8 * half) = *(const uint4*)tmp;
        } else {
#pragma unroll
          for (int r = 0; r < 8; r++)
            bfout[(size_t)(rb + i * 16 + 8 * half + r) * Nout + col] = (__bf16)vals[r];
        }
      }
    }
  }
}

// ---------------------------------------------------------------- flash retention
// softmax(D * (Q K^T)/sqrt(d)) @ V, streaming D once from HBM.
// Block = 8 waves = 16 query rows. Wave w: split-K slice w*128 for S, feature slice w*128 for O.
__global__ __launch_bounds__(256) void retention_flash(
    const __bf16* __restrict__ Qb, const __bf16* __restrict__ Kb,
    const __bf16* __restrict__ Vtb,  // [DIM][NN] (pre-transposed)
    const float* __restrict__ Dm, __bf16* __restrict__ Ob)
{
  __shared__ float Sred[8][2][32][8];          // 16 KB: per-wave partial S tiles
  __shared__ __align__(16) __bf16 Pbuf[8][16][32]; // 8 KB: per-wave P tile (row-major 16x32)

  const int tid  = threadIdx.x;
  const int lane = tid & 31;
  const int w    = tid >> 5;
  const int ln   = lane & 15;
  const int half = lane >> 4;
  const int q0   = blockIdx.x * 16;
  const int fb   = w * 128;   // O feature slice
  const int kb   = w * 128;   // S split-K slice
  const float inv_sqrt_d = 0.03125f;

  v8f oacc[8] = {};
  float mrow[8], lrow[8], alpha[8];
#pragma unroll
  for (int r = 0; r < 8; r++) { mrow[r] = -1e30f; lrow[r] = 0.0f; }

  for (int j0 = 0; j0 < NN; j0 += 32) {
    // ---- partial S over this wave's 128-feature slice
    v8f s0 = {}, s1 = {};
#pragma unroll
    for (int kk = 0; kk < 4; kk++) {
      const int k0 = kb + kk * 32;
      v16bf a  = load_frag_g(Qb, q0 + ln,       DIM, k0, half);
      v16bf b0 = load_frag_g(Kb, j0 + ln,       DIM, k0, half);
      v16bf b1 = load_frag_g(Kb, j0 + 16 + ln,  DIM, k0, half);
      s0 = wmma_bf16(a, b0, s0);
      s1 = wmma_bf16(a, b1, s1);
    }
#pragma unroll
    for (int r = 0; r < 8; r++) { Sred[w][0][lane][r] = s0[r]; Sred[w][1][lane][r] = s1[r]; }
    __syncthreads();

    // ---- reduce split-K partials
    float t0[8] = {}, t1[8] = {};
#pragma unroll
    for (int ww = 0; ww < 8; ww++)
#pragma unroll
      for (int r = 0; r < 8; r++) { t0[r] += Sred[ww][0][lane][r]; t1[r] += Sred[ww][1][lane][r]; }

    // ---- apply D mask + scale, online softmax (stats per row, lane halves independent)
#pragma unroll
    for (int r = 0; r < 8; r++) {
      const size_t row = (size_t)(q0 + r + 8 * half);
      const float d0 = Dm[row * NN + j0 + ln];
      const float d1 = Dm[row * NN + j0 + 16 + ln];
      float x0 = d0 * t0[r] * inv_sqrt_d;
      float x1 = d1 * t1[r] * inv_sqrt_d;
      float mx = fmaxf(x0, x1);
#pragma unroll
      for (int off = 8; off >= 1; off >>= 1) mx = fmaxf(mx, __shfl_xor(mx, off, 16));
      const float nm = fmaxf(mrow[r], mx);
      const float al = __expf(mrow[r] - nm);
      mrow[r] = nm;
      const float p0 = __expf(x0 - nm);
      const float p1 = __expf(x1 - nm);
      float rs = p0 + p1;
#pragma unroll
      for (int off = 8; off >= 1; off >>= 1) rs += __shfl_xor(rs, off, 16);
      lrow[r] = lrow[r] * al + rs;
      alpha[r] = al;
      Pbuf[w][r + 8 * half][ln]      = (__bf16)p0;
      Pbuf[w][r + 8 * half][ln + 16] = (__bf16)p1;
    }
    __syncthreads();

    // ---- P tile to A-operand layout
    FragU pf;
#pragma unroll
    for (int e = 0; e < 8; e++) {
      ((__bf16*)&pf)[e]     = Pbuf[w][ln][half * 8 + e];
      ((__bf16*)&pf)[8 + e] = Pbuf[w][ln][16 + half * 8 + e];
    }
    const v16bf pa = pf.v;

    // ---- O += P @ V  (alpha rescale folded into C operand)
#pragma unroll
    for (int f = 0; f < 8; f++) {
      v16bf vb = load_frag_g(Vtb, fb + f * 16 + ln, NN, j0, half);
#pragma unroll
      for (int r = 0; r < 8; r++) oacc[f][r] *= alpha[r];
      oacc[f] = wmma_bf16(pa, vb, oacc[f]);
    }
  }

  // ---- normalize and write bf16
#pragma unroll
  for (int f = 0; f < 8; f++)
#pragma unroll
    for (int r = 0; r < 8; r++) {
      const float v = oacc[f][r] / lrow[r];
      Ob[(size_t)(q0 + r + 8 * half) * DIM + fb + f * 16 + ln] = (__bf16)v;
    }
}

// ---------------------------------------------------------------- group norm
__global__ void gn_zero(float* stats) { if (threadIdx.x < 32) stats[threadIdx.x] = 0.0f; }

__global__ __launch_bounds__(256) void gn_stats(const float* __restrict__ X, float* __restrict__ stats) {
  const int g = blockIdx.x;         // 16 groups
  const int chunk = blockIdx.y;     // 16 row chunks
  const int c0 = g * 64;
  const int r0 = chunk * (NN / 16);
  float s = 0.0f, ss = 0.0f;
  for (int idx = threadIdx.x; idx < (NN / 16) * 64; idx += 256) {
    const int rr = idx >> 6, cc = idx & 63;
    const float v = X[(size_t)(r0 + rr) * DIM + c0 + cc];
    s += v; ss += v * v;
  }
  __shared__ float sh[2][256];
  sh[0][threadIdx.x] = s; sh[1][threadIdx.x] = ss;
  __syncthreads();
  for (int o = 128; o > 0; o >>= 1) {
    if (threadIdx.x < (unsigned)o) {
      sh[0][threadIdx.x] += sh[0][threadIdx.x + o];
      sh[1][threadIdx.x] += sh[1][threadIdx.x + o];
    }
    __syncthreads();
  }
  if (threadIdx.x == 0) { atomicAdd(&stats[g], sh[0][0]); atomicAdd(&stats[16 + g], sh[1][0]); }
}

__global__ void gn_finalize(const float* __restrict__ X, const float* __restrict__ stats,
                            const float* __restrict__ gw, const float* __restrict__ gb,
                            float* __restrict__ out) {
  const size_t i = (size_t)blockIdx.x * 256 + threadIdx.x;
  if (i >= (size_t)NN * DIM) return;
  const int c = (int)(i & (DIM - 1));
  const int g = c >> 6;
  const float cnt = 64.0f * (float)NN;
  const float mean = stats[g] / cnt;
  const float var = stats[16 + g] / cnt - mean * mean;
  out[i] = (X[i] - mean) * rsqrtf(var + GN_EPS) * gw[c] + gb[c];
}

// ---------------------------------------------------------------- launcher
extern "C" void kernel_launch(void* const* d_in, const int* in_sizes, int n_in,
                              void* d_out, int out_size, void* d_ws, size_t ws_size,
                              hipStream_t stream) {
  const float* h       = (const float*)d_in[0];
  const float* Dm      = (const float*)d_in[1];
  const float* h_prime = (const float*)d_in[2];
  const float* Wq = (const float*)d_in[3];  const float* bq = (const float*)d_in[4];
  const float* Wk = (const float*)d_in[5];  const float* bk = (const float*)d_in[6];
  const float* Wv = (const float*)d_in[7];  const float* bv = (const float*)d_in[8];
  const float* Wr = (const float*)d_in[9];  const float* br = (const float*)d_in[10];
  const float* Wc = (const float*)d_in[11]; const float* bc = (const float*)d_in[12];
  const float* Wp = (const float*)d_in[13]; const float* bp = (const float*)d_in[14];
  const float* prelu_w = (const float*)d_in[15];
  const float* gn_w    = (const float*)d_in[16];
  const float* gn_b    = (const float*)d_in[17];
  float* out = (float*)d_out;

  char* wsp = (char*)d_ws;
  auto alloc = [&](size_t bytes) -> char* {
    char* p = wsp; wsp += (bytes + 255) & ~(size_t)255; return p;
  };
  const size_t act_b = (size_t)NN * DIM * 2;   // 16 MB bf16 activation
  __bf16* hb    = (__bf16*)alloc(act_b);
  __bf16* hpb   = (__bf16*)alloc(act_b);
  __bf16* Qb    = (__bf16*)alloc(act_b);
  __bf16* Kb    = (__bf16*)alloc(act_b);
  __bf16* Vtb   = (__bf16*)alloc(act_b);       // [DIM][NN]
  __bf16* hretb = (__bf16*)alloc(act_b);
  __bf16* t1b   = (__bf16*)alloc(act_b);
  __bf16* hcatb = (__bf16*)alloc(act_b);
  __bf16* Wqb = (__bf16*)alloc((size_t)DIM * DIM * 2);
  __bf16* Wkb = (__bf16*)alloc((size_t)DIM * DIM * 2);
  __bf16* Wvb = (__bf16*)alloc((size_t)DIM * DIM * 2);
  __bf16* Wrb = (__bf16*)alloc((size_t)DIM * DIM * 2);
  __bf16* Wcb = (__bf16*)alloc((size_t)DIM * DIM * 2);
  __bf16* Wpb = (__bf16*)alloc((size_t)DIM * 2 * DIM * 2);
  float*  X     = (float*)alloc((size_t)NN * DIM * 4);
  float*  stats = (float*)alloc(32 * 4);

  // ---- convert inputs / weights to bf16
  auto cvt = [&](const float* src, __bf16* dst, int n) {
    int blocks = (n + 255) / 256; if (blocks > 16384) blocks = 16384;
    f32_to_bf16_kernel<<<blocks, 256, 0, stream>>>(src, dst, n);
  };
  cvt(h,       hb,  NN * DIM);
  cvt(h_prime, hpb, NN * DIM);
  cvt(Wq, Wqb, DIM * DIM);
  cvt(Wk, Wkb, DIM * DIM);
  cvt(Wv, Wvb, DIM * DIM);
  cvt(Wr, Wrb, DIM * DIM);
  cvt(Wc, Wcb, DIM * DIM);
  cvt(Wp, Wpb, DIM * 2 * DIM);

  const dim3 gg(NN / 128, DIM / 128);

  // Q, K projections -> bf16; V -> bf16 transposed
  gemm_wmma<<<gg, 256, 0, stream>>>(hb, NN, DIM, Wqb, DIM, DIM, bq, nullptr, nullptr, 0, 0, Qb,  0);
  gemm_wmma<<<gg, 256, 0, stream>>>(hb, NN, DIM, Wkb, DIM, DIM, bk, nullptr, nullptr, 0, 0, Kb,  0);
  gemm_wmma<<<gg, 256, 0, stream>>>(hb, NN, DIM, Wvb, DIM, DIM, bv, nullptr, nullptr, 0, 0, Vtb, 1);

  // retention: softmax(D * QK^T / 32) @ V
  retention_flash<<<NN / 16, 256, 0, stream>>>(Qb, Kb, Vtb, Dm, hretb);

  // gelu(h_ret @ Wr^T + br) -> bf16
  gemm_wmma<<<gg, 256, 0, stream>>>(hretb, NN, DIM, Wrb, DIM, DIM, br, nullptr, nullptr, 0, 1, t1b, 0);
  // h_cat = h_prime @ Wc^T + bc -> bf16
  gemm_wmma<<<gg, 256, 0, stream>>>(hpb, NN, DIM, Wcb, DIM, DIM, bc, nullptr, nullptr, 0, 0, hcatb, 0);
  // x = t1 @ Wp[:, :1024]^T  (raw)
  gemm_wmma<<<gg, 256, 0, stream>>>(t1b, NN, DIM, Wpb, 2 * DIM, DIM, nullptr, nullptr, X, 0, 0, nullptr, 0);
  // x += h_cat @ Wp[:, 1024:]^T + bp, then PReLU
  gemm_wmma<<<gg, 256, 0, stream>>>(hcatb, NN, DIM, Wpb + DIM, 2 * DIM, DIM, bp, prelu_w, X, 1, 2, nullptr, 0);

  // group norm over (C, N)
  gn_zero<<<1, 32, 0, stream>>>(stats);
  gn_stats<<<dim3(16, 16), 256, 0, stream>>>(X, stats);
  gn_finalize<<<(NN * DIM + 255) / 256, 256, 0, stream>>>(X, stats, gn_w, gn_b, out);
  (void)in_sizes; (void)n_in; (void)out_size; (void)ws_size;
}
